// WideManifold_63891933495803
// MI455X (gfx1250) — compile-verified, matrix operand order checked by reading, not run on hardware
//
#include <hip/hip_runtime.h>
#include <hip/hip_bf16.h>

// ---------------------------------------------------------------------------
// Problem constants (reference: D=512, CELLS=16, BATCH=4, SEQ=1024)
// ---------------------------------------------------------------------------
#define D_DIM   512
#define CELLS   16
#define BATCH   4
#define SEQ     1024
#define BC      (BATCH * CELLS)        // 64
#define MROWS   (BC * SEQ)             // 65536

typedef __attribute__((ext_vector_type(16))) __bf16 v16bf;
typedef __attribute__((ext_vector_type(8)))  float  v8f;

// float -> bf16 bits, round-to-nearest-even
__device__ __forceinline__ unsigned short f2bf(float f) {
    unsigned int u = __float_as_uint(f);
    unsigned int r = u + 0x7FFFu + ((u >> 16) & 1u);
    return (unsigned short)(r >> 16);
}

struct BF16x16 { uint4 lo; uint4 hi; };

// Load a 16x32 bf16 WMMA operand fragment from a row-major [rows, ld] array.
// Per CDNA5 ISA layout: lane L<16 -> row L, K = {0..7, 16..23};
//                       lane L>=16 -> row L-16, K = {8..15, 24..31}.
// `base` points at element [row0][k0]; ld in elements; 16B alignment holds
// whenever k0 % 32 == 0 and ld % 8 == 0.
__device__ __forceinline__ v16bf load_frag(const unsigned short* base, int ld) {
    int lane = threadIdx.x & 31;
    int r    = lane & 15;
    int koff = (lane >> 4) << 3;   // 0 or 8
    const unsigned short* p = base + (size_t)r * ld + koff;
    BF16x16 t;
    t.lo = *(const uint4*)(p);        // K = koff .. koff+7
    t.hi = *(const uint4*)(p + 16);   // K = koff+16 .. koff+23
    return __builtin_bit_cast(v16bf, t);
}

__device__ __forceinline__ v8f wmma_bf16(v16bf a, v16bf b, v8f c) {
    return __builtin_amdgcn_wmma_f32_16x16x32_bf16(
        /*neg_a=*/false, a, /*neg_b=*/false, b,
        /*c_mod=*/(short)0, c, /*reuse_a=*/false, /*reuse_b=*/false);
}

// Wave-level GEMM: acc[MW][NW] (16x16 f32 tiles) += A(16*MW x K) * B^T
// A: row-major [.., lda], rows = M-tile; B: row-major [.., ldb], rows = N-tile
// (i.e. B is supplied as Bt[n][k], so each lane reads contiguous K).
// Runtime K loop on purpose: the loop form keeps VGPR liveness to one K-step
// and lets the backend software-pipeline loads under the WMMAs without
// spilling (full unroll was observed to spill epilogue addresses).
template<int MW, int NW>
__device__ __forceinline__ void wave_gemm(const unsigned short* __restrict__ A, int lda,
                                          const unsigned short* __restrict__ B, int ldb,
                                          int K, v8f (&acc)[MW][NW]) {
    for (int k0 = 0; k0 < K; k0 += 32) {
        // Unconditional speculative prefetch of the next K tile
        // (global_prefetch_b8; translation failures are silently dropped).
        __builtin_prefetch(A + k0 + 32, 0, 1);
        __builtin_prefetch(B + k0 + 32, 0, 1);
        v16bf a[MW], b[NW];
#pragma unroll
        for (int i = 0; i < MW; ++i) a[i] = load_frag(A + (size_t)i * 16 * lda + k0, lda);
#pragma unroll
        for (int j = 0; j < NW; ++j) b[j] = load_frag(B + (size_t)j * 16 * ldb + k0, ldb);
#pragma unroll
        for (int i = 0; i < MW; ++i)
#pragma unroll
            for (int j = 0; j < NW; ++j)
                acc[i][j] = wmma_bf16(a[i], b[j], acc[i][j]);
    }
    // Keep the epilogue (store-address math) out of the accumulation region.
    __builtin_amdgcn_sched_barrier(0);
}

// ---------------------------------------------------------------------------
// Kernel 1: LayerNorm over D=512, emit bf16 nx.  One row per 256-thread block.
// ---------------------------------------------------------------------------
__global__ void ln_kernel(const float* __restrict__ x,
                          const float* __restrict__ gamma,
                          const float* __restrict__ beta,
                          unsigned short* __restrict__ nxb) {
    int row = blockIdx.x;
    const float* xr = x + (size_t)row * D_DIM;
    int t = threadIdx.x;
    float v0 = xr[t], v1 = xr[t + 256];
    float s = v0 + v1;
    float q = v0 * v0 + v1 * v1;
#pragma unroll
    for (int off = 16; off > 0; off >>= 1) {
        s += __shfl_xor(s, off);
        q += __shfl_xor(q, off);
    }
    __shared__ float ss[8], sq[8];
    int w = t >> 5;
    if ((t & 31) == 0) { ss[w] = s; sq[w] = q; }
    __syncthreads();
    float S = 0.f, Q = 0.f;
#pragma unroll
    for (int i = 0; i < 8; ++i) { S += ss[i]; Q += sq[i]; }
    float mu   = S * (1.0f / D_DIM);
    float var  = Q * (1.0f / D_DIM) - mu * mu;
    float rstd = rsqrtf(var + 1e-5f);
    unsigned short* o = nxb + (size_t)row * D_DIM;
    o[t]       = f2bf((v0 - mu) * rstd * gamma[t]       + beta[t]);
    o[t + 256] = f2bf((v1 - mu) * rstd * gamma[t + 256] + beta[t + 256]);
}

// ---------------------------------------------------------------------------
// Kernel 2: fp32 -> bf16 conversion (for Wg)
// ---------------------------------------------------------------------------
__global__ void cvt_bf16_kernel(const float* __restrict__ src,
                                unsigned short* __restrict__ dst, int n) {
    int i = blockIdx.x * blockDim.x + threadIdx.x;
    if (i < n) dst[i] = f2bf(src[i]);
}

// ---------------------------------------------------------------------------
// Kernel 3: qkv = nx @ Wg^T.  M=65536, N=512, K=512.
// Block = 256 thr (8 waves as 4x2), block tile 128x128, wave tile 32x64.
// Writes row-major qkv[bc][t][d] and transposed qkvT[bc][d][t] (both bf16).
// ---------------------------------------------------------------------------
__global__ void gemm1_kernel(const unsigned short* __restrict__ nxb,
                             const unsigned short* __restrict__ wgb,
                             unsigned short* __restrict__ qkv,
                             unsigned short* __restrict__ qkvT) {
    int w  = threadIdx.x >> 5;
    int wm = w >> 1, wn = w & 1;
    int m0 = blockIdx.x * 128 + wm * 32;
    int n0 = blockIdx.y * 128 + wn * 64;

    v8f acc[2][4] = {};
    wave_gemm<2, 4>(nxb + (size_t)m0 * D_DIM, D_DIM,
                    wgb + (size_t)n0 * D_DIM, D_DIM, D_DIM, acc);

    int bc = m0 >> 10;          // 128-row tiles never straddle a bc (1024%128==0)
    int t0 = m0 & 1023;
    unsigned short* qT = qkvT + (size_t)bc * D_DIM * SEQ;
    int lane = threadIdx.x & 31;
    int hi   = lane >> 4;
#pragma unroll
    for (int i = 0; i < 2; ++i)
#pragma unroll
        for (int j = 0; j < 4; ++j) {
            int col   = n0 + j * 16 + (lane & 15);
            int rbase = t0 + i * 16 + hi * 8;
#pragma unroll
            for (int r = 0; r < 8; ++r) {
                unsigned short h = f2bf(acc[i][j][r]);
                int trow = rbase + r;
                qkv[((size_t)bc * SEQ + trow) * D_DIM + col] = h;
                qT[(size_t)col * SEQ + trow] = h;
            }
        }
}

// ---------------------------------------------------------------------------
// Kernel 4: scores + causal mask + softmax -> P (bf16).
// One block per (bc, 16-row q tile); 8 waves each own a 128-col slice, so the
// block holds the full 16x1024 row for exact softmax.
// ---------------------------------------------------------------------------
__global__ void attn_scores_kernel(const unsigned short* __restrict__ qkv,
                                   unsigned short* __restrict__ P) {
    int bc = blockIdx.x;
    int q0 = blockIdx.y * 16;
    int w  = threadIdx.x >> 5;
    int c0 = w * 128;
    int lane = threadIdx.x & 31;
    int hi   = lane >> 4;

    const unsigned short* A  = qkv + ((size_t)bc * SEQ + q0) * D_DIM;
    const unsigned short* Bt = qkv + ((size_t)bc * SEQ + c0) * D_DIM;

    v8f acc[1][8] = {};
    wave_gemm<1, 8>(A, D_DIM, Bt, D_DIM, D_DIM, acc);

    const float scale  = 0.044194173824159216f;   // 512^-0.5
    const float NEGBIG = -3.0e38f;

    __shared__ float red[128];      // 8 waves x 16 rows
    __shared__ float gmax[16], gsum[16];

    float rmax[8];
#pragma unroll
    for (int r = 0; r < 8; ++r) rmax[r] = NEGBIG;
#pragma unroll
    for (int j = 0; j < 8; ++j) {
        int col = c0 + j * 16 + (lane & 15);
#pragma unroll
        for (int r = 0; r < 8; ++r) {
            int row = q0 + hi * 8 + r;
            float s = acc[0][j][r] * scale;
            if (col > row) s = NEGBIG;     // causal mask
            acc[0][j][r] = s;
            rmax[r] = fmaxf(rmax[r], s);
        }
    }
    // reduce across the 16 lanes that share each row (xor 1,2,4,8 stays in half)
#pragma unroll
    for (int r = 0; r < 8; ++r) {
        float v = rmax[r];
        v = fmaxf(v, __shfl_xor(v, 1));
        v = fmaxf(v, __shfl_xor(v, 2));
        v = fmaxf(v, __shfl_xor(v, 4));
        v = fmaxf(v, __shfl_xor(v, 8));
        rmax[r] = v;
    }
    if ((lane & 15) == 0) {
#pragma unroll
        for (int r = 0; r < 8; ++r) red[w * 16 + hi * 8 + r] = rmax[r];
    }
    __syncthreads();
    if (threadIdx.x < 16) {
        float g = NEGBIG;
#pragma unroll
        for (int i = 0; i < 8; ++i) g = fmaxf(g, red[i * 16 + (int)threadIdx.x]);
        gmax[threadIdx.x] = g;
    }
    __syncthreads();

    float rsum[8];
#pragma unroll
    for (int r = 0; r < 8; ++r) rsum[r] = 0.f;
#pragma unroll
    for (int j = 0; j < 8; ++j)
#pragma unroll
        for (int r = 0; r < 8; ++r) {
            float e = __expf(acc[0][j][r] - gmax[hi * 8 + r]);
            acc[0][j][r] = e;
            rsum[r] += e;
        }
#pragma unroll
    for (int r = 0; r < 8; ++r) {
        float v = rsum[r];
        v += __shfl_xor(v, 1);
        v += __shfl_xor(v, 2);
        v += __shfl_xor(v, 4);
        v += __shfl_xor(v, 8);
        rsum[r] = v;
    }
    __syncthreads();               // everyone past the gmax phase before reuse
    if ((lane & 15) == 0) {
#pragma unroll
        for (int r = 0; r < 8; ++r) red[w * 16 + hi * 8 + r] = rsum[r];
    }
    __syncthreads();
    if (threadIdx.x < 16) {
        float g = 0.f;
#pragma unroll
        for (int i = 0; i < 8; ++i) g += red[i * 16 + (int)threadIdx.x];
        gsum[threadIdx.x] = g;
    }
    __syncthreads();

#pragma unroll
    for (int j = 0; j < 8; ++j) {
        int col = c0 + j * 16 + (lane & 15);
#pragma unroll
        for (int r = 0; r < 8; ++r) {
            int row = q0 + hi * 8 + r;
            float p = acc[0][j][r] * (1.0f / gsum[hi * 8 + r]);
            P[((size_t)bc * SEQ + row) * SEQ + col] = f2bf(p);
        }
    }
}

// ---------------------------------------------------------------------------
// Kernel 5: x' = x + P @ V, per bc GEMM 1024x512x1024 (V from qkvT).
// Block tile 128x128 (4x2 waves of 32x64).
// ---------------------------------------------------------------------------
__global__ void attn_out_kernel(const unsigned short* __restrict__ P,
                                const unsigned short* __restrict__ qkvT,
                                const float* __restrict__ x,
                                float* __restrict__ xp) {
    int bc = blockIdx.x;
    int w  = threadIdx.x >> 5;
    int wm = w >> 1, wn = w & 1;
    int m0 = blockIdx.y * 128 + wm * 32;
    int n0 = blockIdx.z * 128 + wn * 64;

    v8f acc[2][4] = {};
    wave_gemm<2, 4>(P    + ((size_t)bc * SEQ + m0) * SEQ, SEQ,
                    qkvT + ((size_t)bc * D_DIM + n0) * SEQ, SEQ, SEQ, acc);

    int lane = threadIdx.x & 31;
    int hi   = lane >> 4;
#pragma unroll
    for (int i = 0; i < 2; ++i)
#pragma unroll
        for (int j = 0; j < 4; ++j) {
            int col   = n0 + j * 16 + (lane & 15);
            int rbase = m0 + i * 16 + hi * 8;
#pragma unroll
            for (int r = 0; r < 8; ++r) {
                size_t idx = ((size_t)bc * SEQ + rbase + r) * D_DIM + col;
                xp[idx] = x[idx] + acc[i][j][r];
            }
        }
}

// ---------------------------------------------------------------------------
// Kernel 6: cell mixing + sin pulse + residual.
// out[b,k,t,d] = x'[b,k,t,d] + sin(sum_c x'[b,c,t,d]*inh[c,k] + phase[k])*rs
// One thread per (b,t,d): 16 cell gathers (L2-resident), 16 outputs.
// ---------------------------------------------------------------------------
__global__ void pulse_kernel(const float* __restrict__ xp,
                             const float* __restrict__ inhibit,
                             const float* __restrict__ phases,
                             const float* __restrict__ res_scale,
                             float* __restrict__ out) {
    __shared__ float inh[CELLS * CELLS];
    __shared__ float ph[CELLS];
    if (threadIdx.x < CELLS * CELLS) inh[threadIdx.x] = inhibit[threadIdx.x];
    if (threadIdx.x < CELLS)         ph[threadIdx.x]  = phases[threadIdx.x];
    __syncthreads();
    float rs = res_scale[0];

    size_t idx = (size_t)blockIdx.x * blockDim.x + threadIdx.x;  // over b*t*d
    int d = (int)(idx & (D_DIM - 1));
    int t = (int)((idx >> 9) & (SEQ - 1));
    int b = (int)(idx >> 19);

    float xv[CELLS];
#pragma unroll
    for (int c = 0; c < CELLS; ++c)
        xv[c] = xp[(((size_t)(b * CELLS + c) * SEQ + t) << 9) + d];

#pragma unroll
    for (int k = 0; k < CELLS; ++k) {
        float s = 0.f;
#pragma unroll
        for (int c = 0; c < CELLS; ++c) s = fmaf(xv[c], inh[c * CELLS + k], s);
        out[(((size_t)(b * CELLS + k) * SEQ + t) << 9) + d] =
            xv[k] + __sinf(s + ph[k]) * rs;
    }
}

// ---------------------------------------------------------------------------
// Host launcher
// ---------------------------------------------------------------------------
extern "C" void kernel_launch(void* const* d_in, const int* in_sizes, int n_in,
                              void* d_out, int out_size, void* d_ws, size_t ws_size,
                              hipStream_t stream) {
    const float* x         = (const float*)d_in[0];
    // d_in[1] = causal mask (bool) — computed analytically, unused
    const float* gamma     = (const float*)d_in[2];
    const float* beta      = (const float*)d_in[3];
    const float* Wg        = (const float*)d_in[4];
    const float* inhibit   = (const float*)d_in[5];
    const float* phases    = (const float*)d_in[6];
    const float* res_scale = (const float*)d_in[7];
    float* out = (float*)d_out;

    char* ws = (char*)d_ws;
    const size_t SZ_NX   = (size_t)MROWS * D_DIM * 2;        //  64 MB bf16 nx
    const size_t SZ_WG   = (size_t)D_DIM * D_DIM * 2;        // 0.5 MB bf16 Wg
    const size_t SZ_QKV  = (size_t)MROWS * D_DIM * 2;        //  64 MB bf16 qkv
    const size_t SZ_QKVT = (size_t)MROWS * D_DIM * 2;        //  64 MB bf16 qkv^T
    const size_t SZ_P    = (size_t)BC * SEQ * SEQ * 2;       // 128 MB bf16 probs

    unsigned short* nxb  = (unsigned short*)(ws);
    unsigned short* wgb  = (unsigned short*)(ws + SZ_NX);
    unsigned short* qkv  = (unsigned short*)(ws + SZ_NX + SZ_WG);
    unsigned short* qkvT = (unsigned short*)(ws + SZ_NX + SZ_WG + SZ_QKV);
    unsigned short* P    = (unsigned short*)(ws + SZ_NX + SZ_WG + SZ_QKV + SZ_QKVT);
    float*          xp   = (float*)(ws + SZ_NX + SZ_WG + SZ_QKV + SZ_QKVT + SZ_P);

    ln_kernel<<<MROWS, 256, 0, stream>>>(x, gamma, beta, nxb);
    cvt_bf16_kernel<<<(D_DIM * D_DIM) / 256, 256, 0, stream>>>(Wg, wgb, D_DIM * D_DIM);
    gemm1_kernel<<<dim3(MROWS / 128, D_DIM / 128), 256, 0, stream>>>(nxb, wgb, qkv, qkvT);
    attn_scores_kernel<<<dim3(BC, SEQ / 16), 256, 0, stream>>>(qkv, P);
    attn_out_kernel<<<dim3(BC, SEQ / 128, D_DIM / 128), 256, 0, stream>>>(P, qkvT, x, xp);
    pulse_kernel<<<(BATCH * SEQ * D_DIM) / 256, 256, 0, stream>>>(xp, inhibit, phases,
                                                                  res_scale, out);
}
